// TransformerLM_15556371546630
// MI455X (gfx1250) — matrix operand order repaired and checked by reading, not compiled, not run on hardware
//
#include <hip/hip_runtime.h>
#include <cstdint>
#include <cstddef>

// ---------------------------------------------------------------------------
// TransformerLM forward for MI455X (gfx1250).
// f16 operands end-to-end for WMMA, f32 accumulate/residual/LN.
// GEMM: 128x128 block tile, K-step 32, double-buffered LDS staged with
// global_load_async_to_lds_b128 (ASYNCcnt), fragments via ds_load_b128.
// ---------------------------------------------------------------------------

typedef _Float16 v16h __attribute__((ext_vector_type(16)));
typedef _Float16 v8h  __attribute__((ext_vector_type(8)));
typedef float    v8f  __attribute__((ext_vector_type(8)));

#define S_LEN   1024
#define BATCH   4
#define DMODEL  1024
#define NHEAD   16
#define HDIM    64
#define NLAYER  2
#define FFDIM   4096
#define MROWS   (S_LEN * BATCH)      // 4096 activation rows

#define USE_ASYNC_LDS 1

static __device__ __forceinline__ v8f wmma_f16(v16h a, v16h b, v8f c) {
  return __builtin_amdgcn_wmma_f32_16x16x32_f16(false, a, false, b, (short)0, c,
                                                false, false);
}
static __device__ __forceinline__ v8f v8f_zero() {
  v8f z = {0.f, 0.f, 0.f, 0.f, 0.f, 0.f, 0.f, 0.f};
  return z;
}
static __device__ __forceinline__ v16h frag_cat(v8h lo, v8h hi) {
  return __builtin_shufflevector(lo, hi, 0, 1, 2, 3, 4, 5, 6, 7,
                                 8, 9, 10, 11, 12, 13, 14, 15);
}

#if USE_ASYNC_LDS
// CDNA5 async global->LDS copy (16B per lane), tracked by ASYNCcnt.
// Generic LDS pointer low 32 bits == wave-relative LDS byte address.
static __device__ __forceinline__ void async_g2l_b128(void* lds, const void* g) {
  const uint32_t loff = (uint32_t)(uintptr_t)lds;
  const uint64_t ga   = (uint64_t)(uintptr_t)g;
  asm volatile("global_load_async_to_lds_b128 %0, %1, off"
               :: "v"(loff), "v"(ga) : "memory");
}
static __device__ __forceinline__ void wait_asynccnt0() {
  asm volatile("s_wait_asynccnt 0x0" ::: "memory");
}
#endif

// ---------------------------------------------------------------------------
// f32 -> f16 conversion (weights once per launch, etc.)
// ---------------------------------------------------------------------------
__global__ __launch_bounds__(256)
void f32_to_f16_kernel(const float* __restrict__ in, _Float16* __restrict__ out,
                       int n) {
  for (int i = blockIdx.x * 256 + threadIdx.x; i < n; i += gridDim.x * 256)
    out[i] = (_Float16)in[i];
}

// ---------------------------------------------------------------------------
// Adaptive input embedding: x[m,:] = table[band][idx] @ proj[band] * 32 + pos[s]
// Writes f32 master (residual path) and f16 copy (GEMM A operand).
// ---------------------------------------------------------------------------
__global__ __launch_bounds__(256)
void adaptive_embed_kernel(const int* __restrict__ tokens,
                           const float* __restrict__ emb0, const float* __restrict__ proj0,
                           const float* __restrict__ emb1, const float* __restrict__ proj1,
                           const float* __restrict__ emb2, const float* __restrict__ proj2,
                           const float* __restrict__ pos,
                           float* __restrict__ x, _Float16* __restrict__ xh) {
  __shared__ float Es[1024];
  const int m = blockIdx.x;           // m = s*B + b
  const int tid = threadIdx.x;
  const int s = m / BATCH;
  const int t = tokens[m];
  const float* tab; const float* pj; int d, idx;
  if (t < 8000)       { tab = emb0; pj = proj0; d = 1024; idx = t;         }
  else if (t < 20000) { tab = emb1; pj = proj1; d = 256;  idx = t - 8000;  }
  else                { tab = emb2; pj = proj2; d = 64;   idx = t - 20000; }
  for (int i = tid; i < d; i += 256) Es[i] = tab[(size_t)idx * d + i];
  __syncthreads();
  #pragma unroll
  for (int i = 0; i < 4; ++i) {
    const int n = tid + i * 256;
    float acc = 0.f;
    for (int k = 0; k < d; ++k)
      acc = fmaf(Es[k], pj[(size_t)k * DMODEL + n], acc);
    const float v = acc * 32.0f + pos[(size_t)s * DMODEL + n];
    x[(size_t)m * DMODEL + n]  = v;
    xh[(size_t)m * DMODEL + n] = (_Float16)v;
  }
}

// ---------------------------------------------------------------------------
// GEMM: C[M,N] = A[M,K](f16) @ W[N,K](f16)^T + bias(f32), optional ReLU.
// Outputs f32 and/or f16 (either pointer may be null).
// ---------------------------------------------------------------------------
__global__ __launch_bounds__(256)
void gemm_h_kernel(const _Float16* __restrict__ A, const _Float16* __restrict__ W,
                   const float* __restrict__ bias,
                   float* __restrict__ C32, _Float16* __restrict__ C16,
                   int N, int K, int relu) {
  __shared__ _Float16 As[2][128][40];  // [buf][m][k], rows 80B (16B aligned)
  __shared__ _Float16 Ws[2][128][40];  // [buf][n][k]
  const int tid  = threadIdx.x;
  const int lane = tid & 31, wv = tid >> 5;
  const int hl   = lane >> 4, mr = lane & 15;
  const int m0 = blockIdx.y * 128, n0 = blockIdx.x * 128;

  v8f acc[8];
  #pragma unroll
  for (int i = 0; i < 8; ++i) acc[i] = v8f_zero();

  auto stage = [&](int buf, int k0) {
    #pragma unroll
    for (int i = 0; i < 2; ++i) {
      const int chunk = tid + i * 256;          // 512 x 16B chunks per matrix
      const int row = chunk >> 2, c = (chunk & 3) << 3;
#if USE_ASYNC_LDS
      async_g2l_b128(&As[buf][row][c], &A[(size_t)(m0 + row) * K + k0 + c]);
      async_g2l_b128(&Ws[buf][row][c], &W[(size_t)(n0 + row) * K + k0 + c]);
#else
      *(v8h*)&As[buf][row][c] = *(const v8h*)&A[(size_t)(m0 + row) * K + k0 + c];
      *(v8h*)&Ws[buf][row][c] = *(const v8h*)&W[(size_t)(n0 + row) * K + k0 + c];
#endif
    }
  };

  stage(0, 0);
  int cur = 0;
  for (int k0 = 0; k0 < K; k0 += 32) {
#if USE_ASYNC_LDS
    wait_asynccnt0();                 // staged tile landed in LDS
#endif
    __syncthreads();
    if (k0 + 32 < K) {
      stage(cur ^ 1, k0 + 32);        // overlap next tile with WMMA chain
      __builtin_prefetch(&A[(size_t)(m0 + (tid & 127)) * K + k0 + 64], 0, 1);
    }

    // A fragment: two contiguous 16B spans -> 2x ds_load_b128
    const _Float16* arow = &As[cur][wv * 16 + mr][0];
    const v16h a = frag_cat(*(const v8h*)(arow + 8 * hl),
                            *(const v8h*)(arow + 16 + 8 * hl));
    #pragma unroll
    for (int nt = 0; nt < 8; ++nt) {
      const _Float16* brow = &Ws[cur][nt * 16 + mr][0];
      const v16h b = frag_cat(*(const v8h*)(brow + 8 * hl),
                              *(const v8h*)(brow + 16 + 8 * hl));
      acc[nt] = wmma_f16(a, b, acc[nt]);
    }
    cur ^= 1;
  }

  #pragma unroll
  for (int nt = 0; nt < 8; ++nt) {
    const int col = n0 + nt * 16 + mr;
    const float bv = bias[col];
    #pragma unroll
    for (int r = 0; r < 8; ++r) {
      const int row = m0 + wv * 16 + r + 8 * hl;   // C frag: M = r + 8*half
      float v = acc[nt][r] + bv;
      if (relu) v = fmaxf(v, 0.f);
      if (C32) C32[(size_t)row * N + col] = v;
      if (C16) C16[(size_t)row * N + col] = (_Float16)v;
    }
  }
}

// ---------------------------------------------------------------------------
// Causal MHA, flash-style online softmax. f16 in (qkv), f16 out.
// grid = (S/128, B*H); block = 256 (8 waves x 16 query rows).
// ---------------------------------------------------------------------------
__global__ __launch_bounds__(256)
void attention_kernel(const _Float16* __restrict__ qkv, _Float16* __restrict__ out) {
  __shared__ _Float16 Qs[8][16][72];   // [wave][m][hd], rows 144B (16B aligned)
  __shared__ _Float16 Ks[32][72];      // [kv col][hd]
  __shared__ _Float16 Vs[64][40];      // [hd][kv]
  __shared__ _Float16 Ps[8][16][40];   // [wave][m][kv] A-frag repack

  const int tid  = threadIdx.x;
  const int lane = tid & 31, wv = tid >> 5;
  const int hl   = lane >> 4, mr = lane & 15;
  const int bh = blockIdx.y, b = bh / NHEAD, hd = bh % NHEAD;
  const int qt0 = blockIdx.x * 128;

  // stage Q (scaled by 64^-0.5) as 16B chunks
  for (int i = tid; i < 1024; i += 256) {
    const int qr = i >> 3, c = (i & 7) << 3;
    const v8h q = *(const v8h*)&qkv[((size_t)((qt0 + qr) * BATCH + b)) * (3 * DMODEL)
                                    + hd * HDIM + c];
    v8h qs;
    #pragma unroll
    for (int e = 0; e < 8; ++e) qs[e] = (_Float16)((float)q[e] * 0.125f);
    *(v8h*)&Qs[qr >> 4][qr & 15][c] = qs;
  }
  __syncthreads();

  v16h aq[2];                          // Q A-frags for the two 32-wide K chunks
  #pragma unroll
  for (int ks = 0; ks < 2; ++ks) {
    const _Float16* qrow = &Qs[wv][mr][ks * 32];
    aq[ks] = frag_cat(*(const v8h*)(qrow + 8 * hl),
                      *(const v8h*)(qrow + 16 + 8 * hl));
  }

  v8f acc[4];
  #pragma unroll
  for (int i = 0; i < 4; ++i) acc[i] = v8f_zero();
  float mrow[8], lrow[8];
  #pragma unroll
  for (int r = 0; r < 8; ++r) { mrow[r] = -1.0e30f; lrow[r] = 0.f; }

  const int nkt = (qt0 + 128) >> 5;    // causal kv-tile bound for this block
  for (int kt = 0; kt < nkt; ++kt) {
    __syncthreads();
    for (int i = tid; i < 256; i += 256) {        // K tile: 256 x 16B chunks
      const int jj = i >> 3, c = (i & 7) << 3;
      *(v8h*)&Ks[jj][c] =
          *(const v8h*)&qkv[((size_t)((kt * 32 + jj) * BATCH + b)) * (3 * DMODEL)
                            + DMODEL + hd * HDIM + c];
    }
    for (int i = tid; i < 2048; i += 256) {       // V tile transposed
      const int jj = i >> 6, d = i & 63;
      Vs[d][jj] = qkv[((size_t)((kt * 32 + jj) * BATCH + b)) * (3 * DMODEL)
                      + 2 * DMODEL + hd * HDIM + d];
    }
    __syncthreads();

    // scores(16x32) = Q(16x64) @ K^T(64x32)
    v8f sc[2];
    #pragma unroll
    for (int nt2 = 0; nt2 < 2; ++nt2) {
      sc[nt2] = v8f_zero();
      #pragma unroll
      for (int ks = 0; ks < 2; ++ks) {
        const _Float16* krow = &Ks[nt2 * 16 + mr][ks * 32];
        const v16h bk = frag_cat(*(const v8h*)(krow + 8 * hl),
                                 *(const v8h*)(krow + 16 + 8 * hl));
        sc[nt2] = wmma_f16(aq[ks], bk, sc[nt2]);
      }
    }

    // online softmax in C-frag layout (16-lane xor-shuffle row reductions)
    #pragma unroll
    for (int r = 0; r < 8; ++r) {
      const int sq = qt0 + wv * 16 + r + 8 * hl;
      float v0 = sc[0][r], v1 = sc[1][r];
      if (kt * 32 + mr > sq)      v0 = -3.0e38f;
      if (kt * 32 + 16 + mr > sq) v1 = -3.0e38f;
      float rm = fmaxf(v0, v1);
      rm = fmaxf(rm, __shfl_xor(rm, 1, 32));
      rm = fmaxf(rm, __shfl_xor(rm, 2, 32));
      rm = fmaxf(rm, __shfl_xor(rm, 4, 32));
      rm = fmaxf(rm, __shfl_xor(rm, 8, 32));
      const float mnew = fmaxf(mrow[r], rm);
      const float rescale = __expf(mrow[r] - mnew);
      const float p0 = __expf(v0 - mnew);
      const float p1 = __expf(v1 - mnew);
      float rs = p0 + p1;
      rs += __shfl_xor(rs, 1, 32);
      rs += __shfl_xor(rs, 2, 32);
      rs += __shfl_xor(rs, 4, 32);
      rs += __shfl_xor(rs, 8, 32);
      lrow[r] = lrow[r] * rescale + rs;
      mrow[r] = mnew;
      #pragma unroll
      for (int nt = 0; nt < 4; ++nt) acc[nt][r] *= rescale;
      Ps[wv][r + 8 * hl][mr]      = (_Float16)p0;
      Ps[wv][r + 8 * hl][16 + mr] = (_Float16)p1;
    }
    __builtin_amdgcn_wave_barrier();   // intra-wave DS write->read ordering

    // acc(16x64) += P(16x32) @ V(32x64)
    const _Float16* prow = &Ps[wv][mr][0];
    const v16h ap = frag_cat(*(const v8h*)(prow + 8 * hl),
                             *(const v8h*)(prow + 16 + 8 * hl));
    #pragma unroll
    for (int nt = 0; nt < 4; ++nt) {
      const _Float16* vrow = &Vs[nt * 16 + mr][0];
      const v16h bv = frag_cat(*(const v8h*)(vrow + 8 * hl),
                               *(const v8h*)(vrow + 16 + 8 * hl));
      acc[nt] = wmma_f16(ap, bv, acc[nt]);
    }
  }

  #pragma unroll
  for (int nt = 0; nt < 4; ++nt)
    #pragma unroll
    for (int r = 0; r < 8; ++r) {
      const int sq = qt0 + wv * 16 + r + 8 * hl;
      out[((size_t)(sq * BATCH + b)) * DMODEL + hd * HDIM + nt * 16 + mr] =
          (_Float16)(acc[nt][r] / lrow[r]);
    }
}

// ---------------------------------------------------------------------------
// out = LayerNorm(x + delta) * w + b ; one block per D=1024 row.
// Emits f32 (residual) and optional f16 (next GEMM A operand).
// ---------------------------------------------------------------------------
__global__ __launch_bounds__(256)
void residual_ln_kernel(const float* __restrict__ x, const float* __restrict__ delta,
                        const float* __restrict__ w, const float* __restrict__ bb,
                        float* __restrict__ out, _Float16* __restrict__ outh) {
  __shared__ float red[256];
  const int row = blockIdx.x, tid = threadIdx.x;
  float v[4];
  float s = 0.f;
  #pragma unroll
  for (int i = 0; i < 4; ++i) {
    const int c = tid + i * 256;
    v[i] = x[(size_t)row * DMODEL + c] + delta[(size_t)row * DMODEL + c];
    s += v[i];
  }
  red[tid] = s; __syncthreads();
  for (int off = 128; off > 0; off >>= 1) {
    if (tid < off) red[tid] += red[tid + off];
    __syncthreads();
  }
  const float mean = red[0] * (1.f / DMODEL);
  __syncthreads();
  s = 0.f;
  #pragma unroll
  for (int i = 0; i < 4; ++i) { const float d = v[i] - mean; s += d * d; }
  red[tid] = s; __syncthreads();
  for (int off = 128; off > 0; off >>= 1) {
    if (tid < off) red[tid] += red[tid + off];
    __syncthreads();
  }
  const float rstd = rsqrtf(red[0] * (1.f / DMODEL) + 1e-5f);
  #pragma unroll
  for (int i = 0; i < 4; ++i) {
    const int c = tid + i * 256;
    const float o = (v[i] - mean) * rstd * w[c] + bb[c];
    out[(size_t)row * DMODEL + c] = o;
    if (outh) outh[(size_t)row * DMODEL + c] = (_Float16)o;
  }
}

// ---------------------------------------------------------------------------
extern "C" void kernel_launch(void* const* d_in, const int* in_sizes, int n_in,
                              void* d_out, int out_size, void* d_ws, size_t ws_size,
                              hipStream_t stream) {
  (void)in_sizes; (void)n_in; (void)out_size; (void)ws_size;

  const int*   tokens = (const int*)  d_in[0];
  // d_in[1] = padding_mask: all-false, ignored.
  const float* emb0   = (const float*)d_in[2];
  const float* proj0  = (const float*)d_in[3];
  const float* emb1   = (const float*)d_in[4];
  const float* proj1  = (const float*)d_in[5];
  const float* emb2   = (const float*)d_in[6];
  const float* proj2  = (const float*)d_in[7];
  const float* pos    = (const float*)d_in[8];
  const float* inW    = (const float*)d_in[9];
  const float* inB    = (const float*)d_in[10];
  const float* outW   = (const float*)d_in[11];
  const float* outB   = (const float*)d_in[12];
  const float* fc1W   = (const float*)d_in[13];
  const float* fc1B   = (const float*)d_in[14];
  const float* fc2W   = (const float*)d_in[15];
  const float* fc2B   = (const float*)d_in[16];
  const float* ln1w   = (const float*)d_in[17];
  const float* ln1b   = (const float*)d_in[18];
  const float* ln2w   = (const float*)d_in[19];
  const float* ln2b   = (const float*)d_in[20];

  // Workspace layout: f32 x | f32 proj | f16 activations | f16 weights (~152 MiB)
  char* p = (char*)d_ws;
  float* xf = (float*)p;        p += (size_t)MROWS * DMODEL * 4;
  float* pr = (float*)p;        p += (size_t)MROWS * DMODEL * 4;
  _Float16* xh    = (_Float16*)p; p += (size_t)MROWS * DMODEL * 2;
  _Float16* qkvh  = (_Float16*)p; p += (size_t)MROWS * 3 * DMODEL * 2;
  _Float16* attnh = (_Float16*)p; p += (size_t)MROWS * DMODEL * 2;
  _Float16* hh    = (_Float16*)p; p += (size_t)MROWS * FFDIM * 2;
  const int nInW  = NLAYER * 3 * DMODEL * DMODEL;
  const int nOutW = NLAYER * DMODEL * DMODEL;
  const int nFc1W = NLAYER * FFDIM * DMODEL;
  const int nFc2W = NLAYER * DMODEL * FFDIM;
  _Float16* inWh  = (_Float16*)p; p += (size_t)nInW * 2;
  _Float16* outWh = (_Float16*)p; p += (size_t)nOutW * 2;
  _Float16* fc1Wh = (_Float16*)p; p += (size_t)nFc1W * 2;
  _Float16* fc2Wh = (_Float16*)p; p += (size_t)nFc2W * 2;

  float* final_out = (float*)d_out;

  // Pre-convert all weights to f16 (both layers at once).
  f32_to_f16_kernel<<<4096, 256, 0, stream>>>(inW,  inWh,  nInW);
  f32_to_f16_kernel<<<4096, 256, 0, stream>>>(outW, outWh, nOutW);
  f32_to_f16_kernel<<<4096, 256, 0, stream>>>(fc1W, fc1Wh, nFc1W);
  f32_to_f16_kernel<<<4096, 256, 0, stream>>>(fc2W, fc2Wh, nFc2W);

  adaptive_embed_kernel<<<MROWS, 256, 0, stream>>>(
      tokens, emb0, proj0, emb1, proj1, emb2, proj2, pos, xf, xh);

  for (int l = 0; l < NLAYER; ++l) {
    // qkv(f16) = x @ in_proj_w^T + b          (4096 x 3072, K=1024)
    gemm_h_kernel<<<dim3(3 * DMODEL / 128, MROWS / 128), 256, 0, stream>>>(
        xh, inWh + (size_t)l * 3 * DMODEL * DMODEL, inB + (size_t)l * 3 * DMODEL,
        nullptr, qkvh, 3 * DMODEL, DMODEL, 0);

    attention_kernel<<<dim3(S_LEN / 128, BATCH * NHEAD), 256, 0, stream>>>(qkvh, attnh);

    // proj(f32) = attn @ out_proj_w^T + b     (4096 x 1024, K=1024)
    gemm_h_kernel<<<dim3(DMODEL / 128, MROWS / 128), 256, 0, stream>>>(
        attnh, outWh + (size_t)l * DMODEL * DMODEL, outB + (size_t)l * DMODEL,
        pr, nullptr, DMODEL, DMODEL, 0);

    // x = LN(x + proj) -> f32 + f16
    residual_ln_kernel<<<MROWS, 256, 0, stream>>>(
        xf, pr, ln1w + (size_t)l * DMODEL, ln1b + (size_t)l * DMODEL, xf, xh);

    // h(f16) = relu(x @ fc1_w^T + b)          (4096 x 4096, K=1024)
    gemm_h_kernel<<<dim3(FFDIM / 128, MROWS / 128), 256, 0, stream>>>(
        xh, fc1Wh + (size_t)l * FFDIM * DMODEL, fc1B + (size_t)l * FFDIM,
        nullptr, hh, FFDIM, DMODEL, 1);

    // proj(f32) = relu(h @ fc2_w^T + b)       (4096 x 1024, K=4096)
    gemm_h_kernel<<<dim3(DMODEL / 128, MROWS / 128), 256, 0, stream>>>(
        hh, fc2Wh + (size_t)l * DMODEL * FFDIM, fc2B + (size_t)l * DMODEL,
        pr, nullptr, DMODEL, FFDIM, 1);

    // x = LN(x + proj); final layer writes d_out (f32), no f16 needed
    const bool last = (l == NLAYER - 1);
    residual_ln_kernel<<<MROWS, 256, 0, stream>>>(
        xf, pr, ln2w + (size_t)l * DMODEL, ln2b + (size_t)l * DMODEL,
        last ? final_out : xf, last ? nullptr : xh);
  }
}
